// GatedDeltaNet_8194797601267
// MI455X (gfx1250) — matrix operand order repaired and checked by reading, not compiled, not run on hardware
//
#include <hip/hip_runtime.h>
#include <hip/hip_bf16.h>

#define DEV __device__ __forceinline__

typedef __attribute__((ext_vector_type(16))) __bf16 v16bf;
typedef __attribute__((ext_vector_type(8)))  float  v8f;
typedef __attribute__((__vector_size__(16))) int    v4i;

struct U4 { unsigned int x, y, z, w; };
union FragU { U4 u[2]; v16bf v; };

DEV unsigned short f2bf(float f) {
  unsigned int u = __float_as_uint(f);
  u += 0x7FFFu + ((u >> 16) & 1u);      // round to nearest even
  return (unsigned short)(u >> 16);
}
DEV float bf2f(unsigned short h) {
  return __uint_as_float(((unsigned int)h) << 16);
}

// ---- CDNA5 async global->LDS path (ASYNCcnt), with sync fallback ----------
#if defined(__has_builtin)
#  if __has_builtin(__builtin_amdgcn_global_load_async_to_lds_b128)
#    define HAS_ASYNC_LDS 1
#  endif
#  if __has_builtin(__builtin_amdgcn_s_wait_asynccnt)
#    define HAS_WAIT_ASYNC 1
#  endif
#endif

typedef __attribute__((address_space(1))) v4i* gv4p;   // global <4 x i32>*
typedef __attribute__((address_space(3))) v4i* lv4p;   // LDS    <4 x i32>*

DEV void wait_async0() {
#if defined(HAS_ASYNC_LDS)
#  if defined(HAS_WAIT_ASYNC)
  __builtin_amdgcn_s_wait_asynccnt(0);
#  else
  asm volatile("s_wait_asynccnt 0x0" ::: "memory");
#  endif
#endif
}

DEV void cp16(const U4* g, U4* l) {
#if defined(HAS_ASYNC_LDS)
  __builtin_amdgcn_global_load_async_to_lds_b128((gv4p)g, (lv4p)l, 0, 0);
#else
  *l = *g;
#endif
}

// ---------------------------------------------------------------------------
// fp32 -> bf16 conversion (grid-stride)
// ---------------------------------------------------------------------------
__global__ __launch_bounds__(256) void f32_to_bf16_kernel(
    const float* __restrict__ in, unsigned short* __restrict__ out, int n) {
  int i = blockIdx.x * 256 + threadIdx.x;
  int stride = gridDim.x * 256;
  for (; i < n; i += stride) out[i] = f2bf(in[i]);
}

// ---------------------------------------------------------------------------
// alpha[t,h] = exp(-exp(A_log[h]) * softplus(x[t]·Wa[h] + dt_bias[h]))
// ---------------------------------------------------------------------------
__global__ __launch_bounds__(256) void alpha_kernel(
    const float* __restrict__ x, const float* __restrict__ Wa,
    const float* __restrict__ dt_bias, const float* __restrict__ A_log,
    float* __restrict__ alpha) {
  int idx = blockIdx.x * 256 + threadIdx.x;   // < 131072
  int t = idx >> 4, h = idx & 15;
  const float* xr = x + (size_t)t * 1024;
  const float* wr = Wa + (size_t)h * 1024;
  float s = 0.f;
  for (int k = 0; k < 1024; k += 4) {
    s = fmaf(xr[k],     wr[k],     s);
    s = fmaf(xr[k + 1], wr[k + 1], s);
    s = fmaf(xr[k + 2], wr[k + 2], s);
    s = fmaf(xr[k + 3], wr[k + 3], s);
  }
  s += dt_bias[h];
  float sp = (s > 20.f) ? s : log1pf(__expf(s));       // softplus
  alpha[idx] = __expf(-__expf(A_log[h]) * sp);
}

// ---------------------------------------------------------------------------
// C[M,N] = A[M,K] * B[N,K]^T  (both bf16, fp32 accumulate via WMMA)
// block tile 128x128, 8 waves, wave tile 32x64 (2x4 frags), K step 32.
// Double-buffered LDS; tile k+1 async-staged (ASYNCcnt) while k is consumed.
// ---------------------------------------------------------------------------
__global__ __launch_bounds__(256) void gemm_bf16_wmma(
    const unsigned short* __restrict__ A,
    const unsigned short* __restrict__ Bm,
    float* __restrict__ Cf, unsigned short* __restrict__ Cb,
    const float* __restrict__ bias, int K, int N) {
  __shared__ U4 lA[2][128 * 5];   // 32 bf16/row data, padded to 5 quads (80 B)
  __shared__ U4 lB[2][128 * 5];

  const int tid  = threadIdx.x;
  const int wave = tid >> 5;
  const int lane = tid & 31;
  const int hi   = (lane >> 4) & 1;
  const int ln   = lane & 15;
  const int wM   = wave & 3;    // 4 wave-rows  x 32
  const int wN   = wave >> 2;   // 2 wave-cols  x 64
  const int blockM = blockIdx.x * 128;
  const int blockN = blockIdx.y * 128;

  v8f acc[2][4];
#pragma unroll
  for (int i = 0; i < 2; ++i)
#pragma unroll
    for (int j = 0; j < 4; ++j)
#pragma unroll
      for (int r = 0; r < 8; ++r) acc[i][j][r] = 0.f;

  const int srow = tid >> 1;        // staging: 2 threads/row, 2 x 16B each
  const int sq   = (tid & 1) * 2;   // quad 0 or 2

  auto stage = [&](int kk, int bufI) {
    const U4* Ag = reinterpret_cast<const U4*>(A + (size_t)(blockM + srow) * K + kk);
    const U4* Bg = reinterpret_cast<const U4*>(Bm + (size_t)(blockN + srow) * K + kk);
    cp16(Ag + sq,     &lA[bufI][srow * 5 + sq]);
    cp16(Ag + sq + 1, &lA[bufI][srow * 5 + sq + 1]);
    cp16(Bg + sq,     &lB[bufI][srow * 5 + sq]);
    cp16(Bg + sq + 1, &lB[bufI][srow * 5 + sq + 1]);
  };

  stage(0, 0);
  wait_async0();
  __syncthreads();

  const int nk = K >> 5;
  for (int kt = 0; kt < nk; ++kt) {
    const int cur = kt & 1;
    if (kt + 1 < nk) stage((kt + 1) << 5, cur ^ 1);

    FragU fa[2], fb[4];
#pragma unroll
    for (int i = 0; i < 2; ++i) {       // A: K chunks at hi*8 and 16+hi*8
      int ml = wM * 32 + i * 16 + ln;
      fa[i].u[0] = lA[cur][ml * 5 + hi];
      fa[i].u[1] = lA[cur][ml * 5 + 2 + hi];
    }
#pragma unroll
    for (int j = 0; j < 4; ++j) {       // B: K chunks at hi*8 and 16+hi*8
      int nl = wN * 64 + j * 16 + ln;
      fb[j].u[0] = lB[cur][nl * 5 + hi];
      fb[j].u[1] = lB[cur][nl * 5 + 2 + hi];
    }
#pragma unroll
    for (int i = 0; i < 2; ++i)
#pragma unroll
      for (int j = 0; j < 4; ++j)
        acc[i][j] = __builtin_amdgcn_wmma_f32_16x16x32_bf16(
            false, fa[i].v, false, fb[j].v, (short)0, acc[i][j], false, false);

    wait_async0();      // own async copies for buf cur^1 complete
    __syncthreads();    // everyone done reading buf cur, staging visible
  }

#pragma unroll
  for (int i = 0; i < 2; ++i) {
#pragma unroll
    for (int j = 0; j < 4; ++j) {
      int n = blockN + wN * 64 + j * 16 + ln;
      float bv = bias ? bias[n] : 0.f;
      int mbase = blockM + wM * 32 + i * 16 + hi * 8;
#pragma unroll
      for (int r = 0; r < 8; ++r) {
        float val = acc[i][j][r] + bv;
        size_t idx = (size_t)(mbase + r) * N + n;
        if (Cf) Cf[idx] = val;
        else    Cb[idx] = f2bf(val);
      }
    }
  }
}

// ---------------------------------------------------------------------------
// Delta-rule scan. One block per (b,h); S[64][64] fp32 in registers:
// thread (rg,e) owns S[rg*16+i][e], i=0..15.  3 barriers per step.
// ---------------------------------------------------------------------------
__global__ __launch_bounds__(256) void scan_kernel(
    const unsigned short* __restrict__ qb, const unsigned short* __restrict__ kb,
    const unsigned short* __restrict__ vb, const unsigned short* __restrict__ bb,
    const float* __restrict__ alpha, float* __restrict__ ys, int Lr) {
  __shared__ float sK[64], sQ[64], sV[64], sB[64], sP1[256], sP2[256], sRed[4];
  const int tid = threadIdx.x;
  const int wave = tid >> 5, lane = tid & 31;
  const int e = tid & 63, rg = tid >> 6;
  const int bh = blockIdx.x;           // 0..63
  const int b = bh >> 4, h = bh & 15;

  float s[16];
#pragma unroll
  for (int i = 0; i < 16; ++i) s[i] = 0.f;

  size_t tbase = ((size_t)b * Lr) * 1024 + (size_t)h * 64;
  for (int l = 0; l < Lr; ++l, tbase += 1024) {
    if (wave == 0) {                           // k + ||k||
      float a0 = bf2f(kb[tbase + lane]), a1 = bf2f(kb[tbase + lane + 32]);
      sK[lane] = a0; sK[lane + 32] = a1;
      float s2 = a0 * a0 + a1 * a1;
      for (int m = 16; m >= 1; m >>= 1) s2 += __shfl_xor(s2, m, 32);
      if (lane == 0) sRed[0] = rsqrtf(s2 + 1e-12f);
    } else if (wave == 1) {                    // q + ||q|| (incl. 1/sqrt(Dh))
      float a0 = bf2f(qb[tbase + lane]), a1 = bf2f(qb[tbase + lane + 32]);
      sQ[lane] = a0; sQ[lane + 32] = a1;
      float s2 = a0 * a0 + a1 * a1;
      for (int m = 16; m >= 1; m >>= 1) s2 += __shfl_xor(s2, m, 32);
      if (lane == 0) sRed[1] = rsqrtf(s2 + 1e-12f) * 0.125f;
    } else if (wave == 2) {                    // v
      sV[lane]      = bf2f(vb[tbase + lane]);
      sV[lane + 32] = bf2f(vb[tbase + lane + 32]);
    } else if (wave == 3) {                    // beta = sigmoid(raw)
      float b0 = bf2f(bb[tbase + lane]), b1 = bf2f(bb[tbase + lane + 32]);
      sB[lane]      = 1.f / (1.f + __expf(-b0));
      sB[lane + 32] = 1.f / (1.f + __expf(-b1));
    } else if (wave == 4 && lane == 0) {
      sRed[2] = alpha[((size_t)b * Lr + l) * 16 + h];
    }
    __syncthreads();                           // B1

    const float a = sRed[2], kninv = sRed[0], qsc = sRed[1];
    float kr[16];
#pragma unroll
    for (int i = 0; i < 16; ++i) kr[i] = sK[rg * 16 + i] * kninv;
    float part = 0.f;
#pragma unroll
    for (int i = 0; i < 16; ++i) { s[i] *= a; part = fmaf(kr[i], s[i], part); }
    sP1[tid] = part;                           // kv partial: k_hat · S
    __syncthreads();                           // B2

    float kv = sP1[e] + sP1[64 + e] + sP1[128 + e] + sP1[192 + e];
    float dlt = (sV[e] - kv) * sB[e];
    float part2 = 0.f;
#pragma unroll
    for (int i = 0; i < 16; ++i) {             // S += k_hat (x) delta ; y = q·S
      s[i] = fmaf(kr[i], dlt, s[i]);
      part2 = fmaf(sQ[rg * 16 + i], s[i], part2);
    }
    sP2[tid] = part2 * qsc;
    __syncthreads();                           // B3
    if (tid < 64)
      ys[tbase + tid] = sP2[tid] + sP2[64 + tid] + sP2[128 + tid] + sP2[192 + tid];
    // no end barrier needed: next step writes sK/sQ/sV/sB/sRed (last read
    // before B3) and sP1/sP2 are rewritten only after B1'/B2'.
  }
}

// ---------------------------------------------------------------------------
// ctx = rmsnorm(ys)*norm_w * silu(gate) -> bf16. One wave32 per (t,h) row.
// ---------------------------------------------------------------------------
__global__ __launch_bounds__(256) void gatenorm_kernel(
    const float* __restrict__ ys, const unsigned short* __restrict__ gb,
    const float* __restrict__ normw, unsigned short* __restrict__ ctxb) {
  const int wave = threadIdx.x >> 5, lane = threadIdx.x & 31;
  const int row = blockIdx.x * 8 + wave;       // (t*16 + h), < 131072
  size_t base = (size_t)row * 64;
  float y0 = ys[base + lane], y1 = ys[base + lane + 32];
  float s2 = y0 * y0 + y1 * y1;
  for (int m = 16; m >= 1; m >>= 1) s2 += __shfl_xor(s2, m, 32);
  float inv = rsqrtf(s2 * (1.f / 64.f) + 1e-6f);
  float g0 = bf2f(gb[base + lane]), g1 = bf2f(gb[base + lane + 32]);
  float c0 = y0 * inv * normw[lane]      * (g0 / (1.f + __expf(-g0)));
  float c1 = y1 * inv * normw[lane + 32] * (g1 / (1.f + __expf(-g1)));
  ctxb[base + lane]      = f2bf(c0);
  ctxb[base + lane + 32] = f2bf(c1);
}

// ---------------------------------------------------------------------------
extern "C" void kernel_launch(void* const* d_in, const int* in_sizes, int n_in,
                              void* d_out, int out_size, void* d_ws, size_t ws_size,
                              hipStream_t stream) {
  (void)in_sizes; (void)n_in; (void)out_size; (void)ws_size;
  const float* x       = (const float*)d_in[0];
  const float* Wq      = (const float*)d_in[1];
  const float* Wk      = (const float*)d_in[2];
  const float* Wv      = (const float*)d_in[3];
  const float* Wg      = (const float*)d_in[4];
  const float* Wb      = (const float*)d_in[5];
  const float* Wa      = (const float*)d_in[6];
  const float* dt_bias = (const float*)d_in[7];
  const float* A_log   = (const float*)d_in[8];
  const float* norm_w  = (const float*)d_in[9];
  const float* Wo      = (const float*)d_in[10];
  const float* bo      = (const float*)d_in[11];

  const size_t M = 8192, D = 1024;     // M = B*L
  char* ws = (char*)d_ws;
  size_t off = 0;
  auto take = [&](size_t bytes) -> char* {
    char* p = ws + off;
    off += (bytes + 255) & ~(size_t)255;
    return p;
  };
  unsigned short* xb   = (unsigned short*)take(M * D * 2);
  unsigned short* wqb  = (unsigned short*)take(D * D * 2);
  unsigned short* wkb  = (unsigned short*)take(D * D * 2);
  unsigned short* wvb  = (unsigned short*)take(D * D * 2);
  unsigned short* wgb  = (unsigned short*)take(D * D * 2);
  unsigned short* wbb  = (unsigned short*)take(D * D * 2);
  unsigned short* wob  = (unsigned short*)take(D * D * 2);
  unsigned short* qb   = (unsigned short*)take(M * D * 2);
  unsigned short* kb   = (unsigned short*)take(M * D * 2);
  unsigned short* vb   = (unsigned short*)take(M * D * 2);
  unsigned short* gb   = (unsigned short*)take(M * D * 2);
  unsigned short* btb  = (unsigned short*)take(M * D * 2);
  float*          alp  = (float*)take(M * 16 * 4);
  float*          ysb  = (float*)take(M * D * 4);
  unsigned short* ctxb = (unsigned short*)take(M * D * 2);

  f32_to_bf16_kernel<<<4096, 256, 0, stream>>>(x,  xb,  (int)(M * D));
  f32_to_bf16_kernel<<<1024, 256, 0, stream>>>(Wq, wqb, (int)(D * D));
  f32_to_bf16_kernel<<<1024, 256, 0, stream>>>(Wk, wkb, (int)(D * D));
  f32_to_bf16_kernel<<<1024, 256, 0, stream>>>(Wv, wvb, (int)(D * D));
  f32_to_bf16_kernel<<<1024, 256, 0, stream>>>(Wg, wgb, (int)(D * D));
  f32_to_bf16_kernel<<<1024, 256, 0, stream>>>(Wb, wbb, (int)(D * D));
  f32_to_bf16_kernel<<<1024, 256, 0, stream>>>(Wo, wob, (int)(D * D));

  alpha_kernel<<<512, 256, 0, stream>>>(x, Wa, dt_bias, A_log, alp);

  dim3 g((unsigned)(M / 128), (unsigned)(D / 128));
  gemm_bf16_wmma<<<g, 256, 0, stream>>>(xb, wqb, nullptr, qb,  nullptr, (int)D, (int)D);
  gemm_bf16_wmma<<<g, 256, 0, stream>>>(xb, wkb, nullptr, kb,  nullptr, (int)D, (int)D);
  gemm_bf16_wmma<<<g, 256, 0, stream>>>(xb, wvb, nullptr, vb,  nullptr, (int)D, (int)D);
  gemm_bf16_wmma<<<g, 256, 0, stream>>>(xb, wgb, nullptr, gb,  nullptr, (int)D, (int)D);
  gemm_bf16_wmma<<<g, 256, 0, stream>>>(xb, wbb, nullptr, btb, nullptr, (int)D, (int)D);

  scan_kernel<<<64, 256, 0, stream>>>(qb, kb, vb, btb, alp, ysb, 2048);

  gatenorm_kernel<<<16384, 256, 0, stream>>>(ysb, gb, norm_w, ctxb);

  gemm_bf16_wmma<<<g, 256, 0, stream>>>(ctxb, wob, (float*)d_out, nullptr, bo, (int)D, (int)D);
}
